// LinearRNNBlock_37039797960730
// MI455X (gfx1250) — matrix operand order repaired and checked by reading, not compiled, not run on hardware
//
#include <hip/hip_runtime.h>
#include <hip/hip_bf16.h>
#include <stdint.h>

typedef __bf16 v16bf __attribute__((ext_vector_type(16)));
typedef float  v8f   __attribute__((ext_vector_type(8)));

#define BM 128
#define BN 128
#define BK 32
#define LDK 40   // BK + 8 halfword padding -> 80B row stride, conflict-free b128 reads

#define USE_ASYNC_LDS 1

__device__ __forceinline__ unsigned short f2bf(float f) {
    unsigned int u = __float_as_uint(f);
    u += 0x7FFFu + ((u >> 16) & 1u);         // round-to-nearest-even
    return (unsigned short)(u >> 16);
}

__device__ __forceinline__ float hsig(float v) {
    // relu6(v+3)/6 - 0.5
    return fminf(fmaxf(v + 3.0f, 0.0f), 6.0f) * (1.0f / 6.0f) - 0.5f;
}

union FragBF {
    v16bf v;
    uint4 q[2];
};

#if USE_ASYNC_LDS
// CDNA5 async global->LDS DMA (16B per lane), tracked by ASYNCcnt.
__device__ __forceinline__ void async_cp16(const unsigned short* g, unsigned short* l) {
    unsigned lds_off = (unsigned)(unsigned long long)
        (__attribute__((address_space(3))) unsigned short*)l;
    asm volatile("global_load_async_to_lds_b128 %0, %1, off"
                 :: "v"(lds_off), "v"(g) : "memory");
}
__device__ __forceinline__ void async_wait0() {
    asm volatile("s_wait_asynccnt 0x0" ::: "memory");
}
#endif

// -------- weight convert + transpose: W[K][N] fp32 -> Wt[N][K] bf16 --------
__global__ __launch_bounds__(256)
void transconv_kernel(const float* __restrict__ W, unsigned short* __restrict__ Wt,
                      int K, int N) {
    int e = blockIdx.x * 256 + threadIdx.x;
    if (e >= K * N) return;
    int k = e / N, n = e % N;
    Wt[(size_t)n * K + k] = f2bf(W[e]);
}

// -------- rmsnorm + hard_sigmoid gate: x row -> fp32 g row --------
__global__ __launch_bounds__(256)
void gate_kernel(const float* __restrict__ x, const float* __restrict__ w,
                 float* __restrict__ g, int C) {
    __shared__ float red[256];
    size_t row = blockIdx.x;
    const float* xr = x + row * C;
    float s = 0.0f;
    for (int i = threadIdx.x; i < C; i += 256) { float v = xr[i]; s += v * v; }
    red[threadIdx.x] = s; __syncthreads();
    for (int off = 128; off > 0; off >>= 1) {
        if (threadIdx.x < off) red[threadIdx.x] += red[threadIdx.x + off];
        __syncthreads();
    }
    float inv = rsqrtf(red[0] / (float)C + 1e-6f);
    float* gr = g + row * C;
    for (int i = threadIdx.x; i < C; i += 256) gr[i] = hsig(xr[i] * inv * w[i]);
}

// -------- rmsnorm of `out` -> bf16 row --------
__global__ __launch_bounds__(256)
void norm2_kernel(const float* __restrict__ o, const float* __restrict__ w,
                  unsigned short* __restrict__ nb, int C) {
    __shared__ float red[256];
    size_t row = blockIdx.x;
    const float* orow = o + row * C;
    float s = 0.0f;
    for (int i = threadIdx.x; i < C; i += 256) { float v = orow[i]; s += v * v; }
    red[threadIdx.x] = s; __syncthreads();
    for (int off = 128; off > 0; off >>= 1) {
        if (threadIdx.x < off) red[threadIdx.x] += red[threadIdx.x + off];
        __syncthreads();
    }
    float inv = rsqrtf(red[0] / (float)C + 1e-6f);
    unsigned short* nr = nb + row * C;
    for (int i = threadIdx.x; i < C; i += 256) nr[i] = f2bf(orow[i] * inv * w[i]);
}

// -------- blocked scan over T (3 phases), emit bf16 state --------
#define NCH 16
#define TCH 256   // T / NCH

// phase 1: per-chunk partial sums -> csum[b][ch][c]
__global__ __launch_bounds__(256)
void chunksum_kernel(const float* __restrict__ g, float* __restrict__ csum,
                     int T, int C) {
    int idx = blockIdx.x * 256 + threadIdx.x;   // 0 .. B*NCH*C-1
    int c = idx % C;
    int r = idx / C;
    int ch = r % NCH, b = r / NCH;
    size_t base = (size_t)b * T * C + (size_t)ch * TCH * C + c;
    float s = 0.0f;
    for (int t = 0; t < TCH; ++t) s += g[base + (size_t)t * C];
    csum[((size_t)b * NCH + ch) * C + c] = s;
}

// phase 2: exclusive scan of the NCH chunk sums per (b,c), in place
__global__ __launch_bounds__(256)
void chunkscan_kernel(float* __restrict__ csum, int C) {
    int idx = blockIdx.x * 256 + threadIdx.x;   // 0 .. B*C-1
    int c = idx % C, b = idx / C;
    float run = 0.0f;
    for (int ch = 0; ch < NCH; ++ch) {
        size_t p = ((size_t)b * NCH + ch) * C + c;
        float v = csum[p];
        csum[p] = run;
        run += v;
    }
}

// phase 3: in-chunk rescan with chunk offset, write bf16 state
__global__ __launch_bounds__(256)
void cumsum2_kernel(const float* __restrict__ g, const float* __restrict__ csum,
                    unsigned short* __restrict__ a1, int T, int C) {
    int idx = blockIdx.x * 256 + threadIdx.x;   // 0 .. B*NCH*C-1
    int c = idx % C;
    int r = idx / C;
    int ch = r % NCH, b = r / NCH;
    size_t base = (size_t)b * T * C + (size_t)ch * TCH * C + c;
    float acc = csum[((size_t)b * NCH + ch) * C + c];
    for (int t = 0; t < TCH; ++t) {
        size_t p = base + (size_t)t * C;
        acc += g[p];
        a1[p] = f2bf(acc);
    }
}

// -------- tile loader: global -> LDS, 16B per thread per matrix, x2 --------
__device__ __forceinline__ void load_tiles(const unsigned short* __restrict__ A,
                                           const unsigned short* __restrict__ Bt,
                                           unsigned short* sA, unsigned short* sB,
                                           int tid, int m0, int n0, int k0, int K) {
    #pragma unroll
    for (int i = 0; i < 2; ++i) {
        int chunk = tid + i * 256;           // 0..511
        int row   = chunk >> 2;              // 0..127
        int col   = (chunk & 3) * 8;         // 0,8,16,24
        const unsigned short* gA = A  + (size_t)(m0 + row) * K + k0 + col;
        const unsigned short* gB = Bt + (size_t)(n0 + row) * K + k0 + col;
        unsigned short* lA = sA + row * LDK + col;
        unsigned short* lB = sB + row * LDK + col;
#if USE_ASYNC_LDS
        async_cp16(gA, lA);
        async_cp16(gB, lB);
#else
        *(uint4*)lA = *(const uint4*)gA;
        *(uint4*)lB = *(const uint4*)gB;
        __builtin_prefetch((const char*)gA + BK * sizeof(unsigned short), 0, 1);
        __builtin_prefetch((const char*)gB + BK * sizeof(unsigned short), 0, 1);
#endif
    }
}

// -------- bf16 WMMA GEMM, double-buffered LDS, fused epilogue --------
// A:  [M][K] bf16 row-major     Bt: [N][K] bf16 (pre-transposed)
// EPI 0: outF = hsig(acc + bias) * extra       (GEMM1, extra = x)
// EPI 1: outB = bf16(relu(acc + bias))         (GEMM2)
// EPI 2: outF = acc + bias + extra             (GEMM3, extra = out residual)
template<int EPI>
__global__ __launch_bounds__(256)
void gemm_bf16_kernel(const unsigned short* __restrict__ A,
                      const unsigned short* __restrict__ Bt,
                      const float* __restrict__ bias,
                      const float* __restrict__ extra,
                      float* __restrict__ outF,
                      unsigned short* __restrict__ outB,
                      int M, int N, int K) {
    __shared__ unsigned short sA[2][BM * LDK];
    __shared__ unsigned short sB[2][BN * LDK];

    const int tid  = threadIdx.x;
    const int lane = tid & 31;
    const int wid  = tid >> 5;
    const int wm   = wid & 3;      // 4 wave-rows of 32
    const int wn   = wid >> 2;     // 2 wave-cols of 64

    const int m0 = blockIdx.y * BM;
    const int n0 = blockIdx.x * BN;

    v8f acc[2][4];
    #pragma unroll
    for (int i = 0; i < 2; ++i)
        #pragma unroll
        for (int j = 0; j < 4; ++j)
            acc[i][j] = (v8f){};

    const int mrow = lane & 15;
    const int aklo = (lane >> 4) * 8;    // ISA A layout: lanes 16-31 -> K+8 / K+24
    const int bklo = (lane >> 4) * 8;    // ISA B layout: lanes 16-31 -> K+8 / K+24

    const int nk = K / BK;

    // prime buffer 0
    load_tiles(A, Bt, sA[0], sB[0], tid, m0, n0, 0, K);
#if USE_ASYNC_LDS
    async_wait0();
#endif
    __syncthreads();

    for (int kt = 0; kt < nk; ++kt) {
        const int cur = kt & 1;
        // kick off async DMA of the next tile into the other buffer
        if (kt + 1 < nk)
            load_tiles(A, Bt, sA[cur ^ 1], sB[cur ^ 1], tid, m0, n0, (kt + 1) * BK, K);

        // fragments from the current buffer
        FragBF af[2], bq[4];
        #pragma unroll
        for (int mt = 0; mt < 2; ++mt) {
            int r = wm * 32 + mt * 16 + mrow;
            af[mt].q[0] = *(const uint4*)(sA[cur] + r * LDK + aklo);
            af[mt].q[1] = *(const uint4*)(sA[cur] + r * LDK + aklo + 16);
        }
        #pragma unroll
        for (int nt = 0; nt < 4; ++nt) {
            int c = wn * 64 + nt * 16 + mrow;
            bq[nt].q[0] = *(const uint4*)(sB[cur] + c * LDK + bklo);
            bq[nt].q[1] = *(const uint4*)(sB[cur] + c * LDK + bklo + 16);
        }
        #pragma unroll
        for (int mt = 0; mt < 2; ++mt)
            #pragma unroll
            for (int nt = 0; nt < 4; ++nt)
                acc[mt][nt] = __builtin_amdgcn_wmma_f32_16x16x32_bf16(
                    false, af[mt].v, false, bq[nt].v,
                    (short)0, acc[mt][nt], false, false);

#if USE_ASYNC_LDS
        async_wait0();       // next tile resident in this wave
#endif
        __syncthreads();     // whole block done reading cur / done writing nxt
    }

    // epilogue: C/D layout -> lane n = lane&15, m = r + 8*(lane>=16)
    const int mBase = m0 + wm * 32 + ((lane >> 4) * 8);
    const int nBase = n0 + wn * 64 + (lane & 15);
    #pragma unroll
    for (int mt = 0; mt < 2; ++mt) {
        #pragma unroll
        for (int nt = 0; nt < 4; ++nt) {
            #pragma unroll
            for (int r = 0; r < 8; ++r) {
                int m = mBase + mt * 16 + r;
                int n = nBase + nt * 16;
                float v = acc[mt][nt][r];
                size_t idx = (size_t)m * N + n;
                if (EPI == 0) {
                    outF[idx] = hsig(v + bias[n]) * extra[idx];
                } else if (EPI == 1) {
                    outB[idx] = f2bf(fmaxf(v + bias[n], 0.0f));
                } else {
                    outF[idx] = v + bias[n] + extra[idx];
                }
            }
        }
    }
}

extern "C" void kernel_launch(void* const* d_in, const int* in_sizes, int n_in,
                              void* d_out, int out_size, void* d_ws, size_t ws_size,
                              hipStream_t stream) {
    const int B = 4, T = 4096, C = 1024, H = 4096;
    const int BT = B * T;   // 16384 = GEMM M

    const float* x   = (const float*)d_in[0];
    const float* n1w = (const float*)d_in[1];
    const float* w1  = (const float*)d_in[2];
    const float* b1  = (const float*)d_in[3];
    const float* n2w = (const float*)d_in[4];
    const float* w2a = (const float*)d_in[5];
    const float* b2a = (const float*)d_in[6];
    const float* w2b = (const float*)d_in[7];
    const float* b2b = (const float*)d_in[8];
    float* y = (float*)d_out;

    char* ws = (char*)d_ws;
    size_t off = 0;
    auto alloc = [&](size_t bytes) -> char* {
        char* p = ws + off;
        off = (off + bytes + 255) & ~(size_t)255;
        return p;
    };
    float*          outF = (float*)alloc((size_t)BT * C * 4);          // 64 MB
    float*          gbuf = (float*)alloc((size_t)BT * C * 4);          // 64 MB
    unsigned short* a1   = (unsigned short*)alloc((size_t)BT * C * 2); // 32 MB
    unsigned short* n2b  = (unsigned short*)alloc((size_t)BT * C * 2); // 32 MB
    unsigned short* hb   = (unsigned short*)alloc((size_t)BT * H * 2); // 128 MB
    unsigned short* w1t  = (unsigned short*)alloc((size_t)C * C * 2);
    unsigned short* w2at = (unsigned short*)alloc((size_t)C * H * 2);
    unsigned short* w2bt = (unsigned short*)alloc((size_t)H * C * 2);
    float*          csum = (float*)alloc((size_t)B * NCH * C * 4);     // 256 KB

    // weights -> bf16 [N][K]
    transconv_kernel<<<(C * C + 255) / 256, 256, 0, stream>>>(w1, w1t, C, C);
    transconv_kernel<<<(C * H + 255) / 256, 256, 0, stream>>>(w2a, w2at, C, H);
    transconv_kernel<<<(H * C + 255) / 256, 256, 0, stream>>>(w2b, w2bt, H, C);

    // gate = hsig(rmsnorm(x, n1w)); blocked prefix-sum over T -> bf16 state
    gate_kernel<<<BT, 256, 0, stream>>>(x, n1w, gbuf, C);
    chunksum_kernel<<<(B * NCH * C) / 256, 256, 0, stream>>>(gbuf, csum, T, C);
    chunkscan_kernel<<<(B * C) / 256, 256, 0, stream>>>(csum, C);
    cumsum2_kernel<<<(B * NCH * C) / 256, 256, 0, stream>>>(gbuf, csum, a1, T, C);

    // out = hsig(state @ W1 + b1) * x
    gemm_bf16_kernel<0><<<dim3(C / BN, BT / BM), 256, 0, stream>>>(
        a1, w1t, b1, x, outF, nullptr, BT, C, C);

    // rmsnorm(out, n2w) -> bf16
    norm2_kernel<<<BT, 256, 0, stream>>>(outF, n2w, n2b, C);

    // h = relu(norm2 @ W2a + b2a) -> bf16
    gemm_bf16_kernel<1><<<dim3(H / BN, BT / BM), 256, 0, stream>>>(
        n2b, w2at, b2a, nullptr, nullptr, hb, BT, H, C);

    // y = h @ W2b + b2b + out
    gemm_bf16_kernel<2><<<dim3(C / BN, BT / BM), 256, 0, stream>>>(
        hb, w2bt, b2b, outF, y, nullptr, BT, C, H);
}
